// GeoVote_9096740733300
// MI455X (gfx1250) — compile-verified, matrix operand order checked by reading, not run on hardware
//
#include <hip/hip_runtime.h>
#include <hip/hip_bf16.h>

// ---------------------------------------------------------------------------
// CDNA5 WMMA types (wave32): 16x16x32 bf16 -> f32 accumulate
// ---------------------------------------------------------------------------
typedef __attribute__((ext_vector_type(16))) __bf16 v16bf;
typedef __attribute__((ext_vector_type(8)))  __bf16 v8bf;
typedef __attribute__((ext_vector_type(8)))  float  v8f;

__device__ __forceinline__ __bf16 f2bf(float f) {
    return (__bf16)f;     // native v_cvt_*_bf16_f32 path on gfx1250
}

static __host__ __device__ inline int ceil_div(int a, int b) { return (a + b - 1) / b; }
static inline int pad64(int x) { return (x + 63) & ~63; }

// ---------------------------------------------------------------------------
// Farthest point sampling: one workgroup per batch
// ---------------------------------------------------------------------------
__global__ __launch_bounds__(256)
void fps_kernel(const float* __restrict__ xyz, int Nq, int npoint,
                int* __restrict__ out, float* __restrict__ dist) {
    const int b = blockIdx.x;
    const float* P = xyz + (size_t)b * Nq * 3;
    float* D = dist + (size_t)b * Nq;
    __shared__ float s_val[256];
    __shared__ int   s_idx[256];
    __shared__ int   s_far;
    const int tid = threadIdx.x;

    for (int n = tid; n < Nq; n += 256) D[n] = 1e10f;
    if (tid == 0) s_far = 0;
    __syncthreads();

    for (int i = 0; i < npoint; ++i) {
        const int far = s_far;
        if (tid == 0) out[b * npoint + i] = far;
        const float cx = P[far * 3 + 0], cy = P[far * 3 + 1], cz = P[far * 3 + 2];
        float bv = -1.0f; int bi = 0;
        for (int n = tid; n < Nq; n += 256) {
            float dx = P[n * 3 + 0] - cx;
            float dy = P[n * 3 + 1] - cy;
            float dz = P[n * 3 + 2] - cz;
            float d  = dx * dx + dy * dy + dz * dz;
            float nd = fminf(D[n], d);
            D[n] = nd;
            if (nd > bv) { bv = nd; bi = n; }
        }
        s_val[tid] = bv; s_idx[tid] = bi;
        __syncthreads();
        for (int s = 128; s > 0; s >>= 1) {
            if (tid < s && s_val[tid + s] > s_val[tid]) {
                s_val[tid] = s_val[tid + s]; s_idx[tid] = s_idx[tid + s];
            }
            __syncthreads();
        }
        if (tid == 0) s_far = s_idx[0];
        __syncthreads();
    }
}

__global__ void gather_xyz_kernel(const float* __restrict__ xs, const int* __restrict__ idx,
                                  int Nq, int npoint, int Btot, float* __restrict__ out) {
    int t = blockIdx.x * blockDim.x + threadIdx.x;
    if (t >= Btot * npoint) return;
    int b = t / npoint;
    int j = idx[t];
    const float* p = xs + ((size_t)b * Nq + j) * 3;
    out[t * 3 + 0] = p[0];
    out[t * 3 + 1] = p[1];
    out[t * 3 + 2] = p[2];
}

// ---------------------------------------------------------------------------
// Ball query: first nsample in-radius indices in index order, pad with first
// ---------------------------------------------------------------------------
__global__ __launch_bounds__(128)
void ball_query_kernel(const float* __restrict__ xs, const float* __restrict__ new_xyz,
                       int Nq, int npoint, int nsample, float r2,
                       int* __restrict__ idx_out) {
    const int bp = blockIdx.x;
    const int b  = bp / npoint;
    const float* P = xs + (size_t)b * Nq * 3;
    const float cx = new_xyz[(size_t)bp * 3 + 0];
    const float cy = new_xyz[(size_t)bp * 3 + 1];
    const float cz = new_xyz[(size_t)bp * 3 + 2];

    __shared__ int s_cnt;
    __shared__ int s_wcnt[4];
    __shared__ int s_out[32];
    const int tid  = threadIdx.x;
    const int lane = tid & 31;
    const int wv   = tid >> 5;
    if (tid == 0) s_cnt = 0;
    __syncthreads();

    for (int base = 0; base < Nq; base += 128) {
        const int n = base + tid;
        bool in = false;
        if (n < Nq) {
            float dx = P[n * 3 + 0] - cx;
            float dy = P[n * 3 + 1] - cy;
            float dz = P[n * 3 + 2] - cz;
            in = (dx * dx + dy * dy + dz * dz) <= r2;
        }
        unsigned mask = (unsigned)__ballot(in);
        if (lane == 0) s_wcnt[wv] = __popc(mask);
        __syncthreads();
        int before = s_cnt;
        for (int w = 0; w < wv; ++w) before += s_wcnt[w];
        int pos = before + __popc(mask & ((1u << lane) - 1u));
        if (in && pos < nsample) s_out[pos] = n;
        __syncthreads();
        if (tid == 0) s_cnt += s_wcnt[0] + s_wcnt[1] + s_wcnt[2] + s_wcnt[3];
        __syncthreads();
        if (s_cnt >= nsample) break;
    }
    if (tid == 0) {
        int c = s_cnt;
        if (c < 1) { s_out[0] = 0; c = 1; }
        if (c > nsample) c = nsample;
        int first = s_out[0];
        for (int s = c; s < nsample; ++s) s_out[s] = first;
    }
    __syncthreads();
    if (tid < nsample) idx_out[(size_t)bp * nsample + tid] = s_out[tid];
}

// ---------------------------------------------------------------------------
// Grouping into K-padded rows (zero-filled pad region)
// ---------------------------------------------------------------------------
__global__ void group_kernel(const float* __restrict__ xs, const float* __restrict__ fs,
                             const float* __restrict__ new_xyz, const int* __restrict__ idx,
                             int Nq, int npoint, int nsample, int Cf, int Kp, int total,
                             float* __restrict__ X) {
    int t = blockIdx.x * blockDim.x + threadIdx.x;
    if (t >= total) return;
    int p  = (t / nsample) % npoint;
    int b  = t / (nsample * npoint);
    int j  = idx[t];
    if (j > Nq - 1) j = Nq - 1;
    const float* ps = xs + ((size_t)b * Nq + j) * 3;
    const float* c  = new_xyz + ((size_t)b * npoint + p) * 3;
    float* row = X + (size_t)t * Kp;
    row[0] = ps[0] - c[0];
    row[1] = ps[1] - c[1];
    row[2] = ps[2] - c[2];
    const float* f = fs + ((size_t)b * Nq + j) * Cf;
    for (int ci = 0; ci < Cf; ++ci) row[3 + ci] = f[ci];
    for (int ci = 3 + Cf; ci < Kp; ++ci) row[ci] = 0.0f;
}

// ---------------------------------------------------------------------------
// Weight pack: fp32 [N][K] -> bf16 [Np][Kp], zero-filled
// ---------------------------------------------------------------------------
__global__ void pack_w_kernel(const float* __restrict__ W, int N, int K, int Kp, int Np,
                              __bf16* __restrict__ out) {
    int t = blockIdx.x * blockDim.x + threadIdx.x;
    if (t >= Np * Kp) return;
    int n = t / Kp, k = t % Kp;
    float f = (n < N && k < K) ? W[(size_t)n * K + k] : 0.0f;
    out[t] = f2bf(f);
}

// ---------------------------------------------------------------------------
// WMMA GEMM:  Y[M,N] = A[M,Ka] * Wp[Np,Ka]^T + bias
//   A: fp32, K-padded (Ka % 64 == 0, zero pad), M % 64 == 0
//   Wp: bf16 pre-packed, Np % 64 == 0
// Block tile 64x64x64: 128 threads = 4 waves (2x2), each wave owns a 2x2 set
// of 16x16 WMMA tiles -> 8 v_wmma per wave per K-tile.
// A staged via b128 loads + native bf16 convert + ds_store_b128;
// B staged via CDNA5 async global->LDS b128 copies (ASYNCcnt).
// ---------------------------------------------------------------------------
__global__ __launch_bounds__(128)
void gemm_bias_kernel(const float* __restrict__ A, int Ka,
                      const __bf16* __restrict__ Wp,
                      const float* __restrict__ bias,
                      float* __restrict__ Y, int strideY,
                      int M, int N) {
    __shared__ __bf16 As[64][72];   // 144B row stride: 16B aligned, bank-spread
    __shared__ __bf16 Bs[64][72];

    const int tid   = threadIdx.x;
    const int wave  = tid >> 5;
    const int lane  = tid & 31;
    const int half  = lane >> 4;
    const int lm    = lane & 15;
    const int waveM = wave & 1;
    const int waveN = wave >> 1;
    const size_t mBase = (size_t)blockIdx.x * 64;
    const size_t nBase = (size_t)blockIdx.y * 64;

    const int sr = tid >> 1;            // staging row 0..63
    const int sc = (tid & 1) * 32;      // staging col base (elements)

    v8f acc00 = {0.f,0.f,0.f,0.f,0.f,0.f,0.f,0.f};
    v8f acc01 = acc00, acc10 = acc00, acc11 = acc00;

    for (int k0 = 0; k0 < Ka; k0 += 64) {
        // ---- B tile: bf16 weights, async copy global -> LDS (b128 x4) ----
        {
            const __bf16* g = Wp + (nBase + sr) * (size_t)Ka + k0 + sc;
            #pragma unroll
            for (int i = 0; i < 4; ++i) {
                unsigned l = (unsigned)(uintptr_t)&Bs[sr][sc + 8 * i];
                asm volatile("global_load_async_to_lds_b128 %0, %1, off"
                             :: "v"(l), "v"(g + 8 * i) : "memory");
            }
        }
        // ---- A tile: fp32 -> bf16 (native converts), vectorized ----
        {
            const float4* g = (const float4*)(A + (mBase + sr) * (size_t)Ka + k0 + sc);
            #pragma unroll
            for (int i = 0; i < 4; ++i) {
                float4 f0 = g[2 * i], f1 = g[2 * i + 1];
                v8bf h;
                h[0] = f2bf(f0.x); h[1] = f2bf(f0.y); h[2] = f2bf(f0.z); h[3] = f2bf(f0.w);
                h[4] = f2bf(f1.x); h[5] = f2bf(f1.y); h[6] = f2bf(f1.z); h[7] = f2bf(f1.w);
                *(v8bf*)&As[sr][sc + 8 * i] = h;
            }
        }
        if (k0 + 64 < Ka)
            __builtin_prefetch(A + (mBase + sr) * (size_t)Ka + k0 + 64 + sc, 0, 1);

        asm volatile("s_wait_asynccnt 0" ::: "memory");
        __syncthreads();

        // ---- 2x2 WMMA tiles per wave, two K-halves -> 8 WMMAs ----
        const int ra0 = 32 * waveM + lm, ra1 = ra0 + 16;
        const int rb0 = 32 * waveN + lm, rb1 = rb0 + 16;
        #pragma unroll
        for (int ks = 0; ks < 64; ks += 32) {
            v8bf alo0 = *(const v8bf*)&As[ra0][ks + half * 8];
            v8bf ahi0 = *(const v8bf*)&As[ra0][ks + 16 + half * 8];
            v8bf alo1 = *(const v8bf*)&As[ra1][ks + half * 8];
            v8bf ahi1 = *(const v8bf*)&As[ra1][ks + 16 + half * 8];
            v8bf blo0 = *(const v8bf*)&Bs[rb0][ks + half * 8];
            v8bf bhi0 = *(const v8bf*)&Bs[rb0][ks + 16 + half * 8];
            v8bf blo1 = *(const v8bf*)&Bs[rb1][ks + half * 8];
            v8bf bhi1 = *(const v8bf*)&Bs[rb1][ks + 16 + half * 8];
            v16bf a0 = __builtin_shufflevector(alo0, ahi0, 0,1,2,3,4,5,6,7,8,9,10,11,12,13,14,15);
            v16bf a1 = __builtin_shufflevector(alo1, ahi1, 0,1,2,3,4,5,6,7,8,9,10,11,12,13,14,15);
            v16bf b0 = __builtin_shufflevector(blo0, bhi0, 0,1,2,3,4,5,6,7,8,9,10,11,12,13,14,15);
            v16bf b1 = __builtin_shufflevector(blo1, bhi1, 0,1,2,3,4,5,6,7,8,9,10,11,12,13,14,15);
            acc00 = __builtin_amdgcn_wmma_f32_16x16x32_bf16(false, a0, false, b0, (short)0, acc00, false, false);
            acc01 = __builtin_amdgcn_wmma_f32_16x16x32_bf16(false, a0, false, b1, (short)0, acc01, false, false);
            acc10 = __builtin_amdgcn_wmma_f32_16x16x32_bf16(false, a1, false, b0, (short)0, acc10, false, false);
            acc11 = __builtin_amdgcn_wmma_f32_16x16x32_bf16(false, a1, false, b1, (short)0, acc11, false, false);
        }
        __syncthreads();
    }

    // ---- store: D layout -> VGPR v: M = v + 8*half, N = lane%16 ----
    auto store_tile = [&](const v8f& acc, size_t mT, int nT) {
        const int n = nT + lm;
        if (n < N) {
            const float bn = bias[n];
            #pragma unroll
            for (int v = 0; v < 8; ++v) {
                size_t m = mT + v + 8 * half;         // M % 64 == 0 -> in range
                Y[m * (size_t)strideY + n] = acc[v] + bn;
            }
        }
    };
    const size_t mT = mBase + 32 * waveM;
    const int    nT = (int)nBase + 32 * waveN;
    store_tile(acc00, mT,      nT);
    store_tile(acc01, mT,      nT + 16);
    store_tile(acc10, mT + 16, nT);
    store_tile(acc11, mT + 16, nT + 16);
}

// ---------------------------------------------------------------------------
// BatchNorm stats / normalize (channel-padded activations, stride Cp)
// ---------------------------------------------------------------------------
__global__ void zero_kernel(float* __restrict__ p, int n) {
    int t = blockIdx.x * blockDim.x + threadIdx.x;
    if (t < n) p[t] = 0.0f;
}

__global__ __launch_bounds__(256)
void bn_stats_kernel(const float* __restrict__ Y, long long M, int C, int Cp,
                     float* __restrict__ stats) {
    extern __shared__ float sh[];       // 2*C floats
    for (int i = threadIdx.x; i < 2 * C; i += blockDim.x) sh[i] = 0.0f;
    __syncthreads();
    long long total = M * (long long)C;
    for (long long t = (long long)blockIdx.x * blockDim.x + threadIdx.x; t < total;
         t += (long long)gridDim.x * blockDim.x) {
        int c = (int)(t % C);
        long long m = t / C;
        float v = Y[m * Cp + c];
        atomicAdd(&sh[c], v);
        atomicAdd(&sh[C + c], v * v);
    }
    __syncthreads();
    for (int i = threadIdx.x; i < 2 * C; i += blockDim.x) atomicAdd(&stats[i], sh[i]);
}

// Normalize + ReLU in place; zero-fills pad channels (c >= C) so the buffer is
// directly usable as the next GEMM's zero-padded K input.
__global__ __launch_bounds__(256)
void bn_relu_kernel(float* __restrict__ Y, long long M, int C, int Cp,
                    const float* __restrict__ stats, const float* __restrict__ g,
                    const float* __restrict__ beta) {
    long long total = M * (long long)Cp;
    float invM = 1.0f / (float)M;
    for (long long t = (long long)blockIdx.x * blockDim.x + threadIdx.x; t < total;
         t += (long long)gridDim.x * blockDim.x) {
        int c = (int)(t % Cp);
        if (c < C) {
            float mean = stats[c] * invM;
            float var  = stats[C + c] * invM - mean * mean;
            float x = Y[t];
            x = g[c] * (x - mean) * rsqrtf(var + 1e-5f) + beta[c];
            Y[t] = fmaxf(x, 0.0f);
        } else {
            Y[t] = 0.0f;
        }
    }
}

// ---------------------------------------------------------------------------
// Max-pool over nsample (input stride Cp, tight fp32 output)
// ---------------------------------------------------------------------------
__global__ void maxpool_kernel(const float* __restrict__ X, int rows, int nsample,
                               int C, int Cp, float* __restrict__ out) {
    int t = blockIdx.x * blockDim.x + threadIdx.x;
    int total = rows * C;
    if (t >= total) return;
    int r = t / C, c = t % C;
    const float* p = X + (size_t)r * nsample * Cp + c;
    float m = p[0];
    for (int s = 1; s < nsample; ++s) m = fmaxf(m, p[(size_t)s * Cp]);
    out[t] = m;
}

// ---------------------------------------------------------------------------
// 3-NN inverse-distance interpolation into fused feature buffer
// ---------------------------------------------------------------------------
__global__ __launch_bounds__(256)
void upsample_kernel(const float* __restrict__ tgt_xyz, const float* __restrict__ src_xyz,
                     const float* __restrict__ src_feat, int N, int np, int Cs,
                     float* __restrict__ fused, int Ctot, int coff) {
    const int b = blockIdx.y;
    extern __shared__ float sp[];       // np*3 floats
    for (int i = threadIdx.x; i < np * 3; i += blockDim.x)
        sp[i] = src_xyz[(size_t)b * np * 3 + i];
    __syncthreads();

    int n = blockIdx.x * blockDim.x + threadIdx.x;
    if (n >= N) return;
    const float* q = tgt_xyz + ((size_t)b * N + n) * 3;
    const float qx = q[0], qy = q[1], qz = q[2];
    float d0 = 1e30f, d1 = 1e30f, d2 = 1e30f;
    int   i0 = 0,     i1 = 0,     i2 = 0;
    for (int j = 0; j < np; ++j) {
        float dx = sp[j * 3 + 0] - qx;
        float dy = sp[j * 3 + 1] - qy;
        float dz = sp[j * 3 + 2] - qz;
        float d  = dx * dx + dy * dy + dz * dz;
        if (d < d0)      { d2 = d1; i2 = i1; d1 = d0; i1 = i0; d0 = d; i0 = j; }
        else if (d < d1) { d2 = d1; i2 = i1; d1 = d;  i1 = j; }
        else if (d < d2) { d2 = d;  i2 = j; }
    }
    float w0 = 1.0f / (d0 + 1e-8f);
    float w1 = 1.0f / (d1 + 1e-8f);
    float w2 = 1.0f / (d2 + 1e-8f);
    float ws = w0 + w1 + w2;
    w0 /= ws; w1 /= ws; w2 /= ws;
    const float* F = src_feat + (size_t)b * np * Cs;
    float* o = fused + ((size_t)b * N + n) * Ctot + coff;
    for (int c = 0; c < Cs; ++c)
        o[c] = w0 * F[(size_t)i0 * Cs + c] + w1 * F[(size_t)i1 * Cs + c]
             + w2 * F[(size_t)i2 * Cs + c];
}

// ---------------------------------------------------------------------------
// Host-side orchestration
// ---------------------------------------------------------------------------
extern "C" void kernel_launch(void* const* d_in, const int* in_sizes, int n_in,
                              void* d_out, int out_size, void* d_ws, size_t ws_size,
                              hipStream_t stream) {
    (void)in_sizes; (void)n_in; (void)out_size; (void)ws_size;

    const int Bc = 4, Nc = 16384, NS = 32;
    const int sa_np[4]  = {1024, 256, 64, 16};
    const float sa_r[4] = {0.01f, 0.02f, 0.04f, 0.08f};
    const int sa_mlp[4][3] = {{32,32,64},{64,64,128},{128,128,256},{256,256,512}};

    // ---- input unpack (insertion order) ----
    int p = 0;
    const float* xyz = (const float*)d_in[p++];
    const float *saw[4][3], *sab[4][3], *sag[4][3], *sabt[4][3];
    for (int l = 0; l < 4; ++l)
        for (int j = 0; j < 3; ++j) {
            saw[l][j]  = (const float*)d_in[p++];
            sab[l][j]  = (const float*)d_in[p++];
            sag[l][j]  = (const float*)d_in[p++];
            sabt[l][j] = (const float*)d_in[p++];
        }
    const float *hw[4], *hb[4], *hg[4] = {0,0,0,0}, *hbt[4] = {0,0,0,0};
    for (int i = 0; i < 4; ++i) {
        hw[i] = (const float*)d_in[p++];
        hb[i] = (const float*)d_in[p++];
        if (i < 3) { hg[i] = (const float*)d_in[p++]; hbt[i] = (const float*)d_in[p++]; }
    }

    // ---- workspace carve ----
    char* wsp = (char*)d_ws;
    auto walloc = [&](size_t bytes) -> void* {
        void* r = (void*)wsp;
        wsp += (bytes + 255) & ~(size_t)255;
        return r;
    };
    float* fps_dist = (float*)walloc((size_t)Bc * Nc * 4);
    int*   fps_idx  = (int*)  walloc((size_t)Bc * 1024 * 4);
    float* nxyz[4];  for (int l = 0; l < 4; ++l) nxyz[l]  = (float*)walloc((size_t)Bc * sa_np[l] * 3 * 4);
    float* fslvl[4]; for (int l = 0; l < 4; ++l) fslvl[l] = (float*)walloc((size_t)Bc * sa_np[l] * sa_mlp[l][2] * 4);
    int*    ball  = (int*)   walloc((size_t)Bc * 1024 * NS * 4);
    float*  gbuf  = (float*) walloc((size_t)Bc * 1024 * NS * 64 * 4);     // max M*Kp grouped
    float*  stats = (float*) walloc(1024 * 4);
    __bf16* wpack = (__bf16*)walloc((size_t)512 * 960 * 2);               // max packed weight
    float*  actA  = (float*) walloc((size_t)Bc * Nc * 512 * 4);           // even outputs / h0 / h2
    float*  actB  = (float*) walloc((size_t)Bc * Nc * 256 * 4);           // odd outputs / h1

    auto run_gemm = [&](const float* A, int Ka, const float* W, int N, int K,
                        const float* bias, float* Y, int strideY, int M) {
        const int Np = ceil_div(N, 64) * 64;
        const int tot = Np * Ka;
        pack_w_kernel<<<ceil_div(tot, 256), 256, 0, stream>>>(W, N, K, Ka, Np, wpack);
        dim3 grid(M / 64, Np / 64);
        gemm_bias_kernel<<<grid, 128, 0, stream>>>(A, Ka, wpack, bias, Y, strideY, M, N);
    };
    auto run_bn = [&](float* Y, long long M, int C, int Cp, const float* g, const float* beta) {
        zero_kernel<<<ceil_div(2 * C, 256), 256, 0, stream>>>(stats, 2 * C);
        bn_stats_kernel<<<512, 256, (size_t)2 * C * 4, stream>>>(Y, M, C, Cp, stats);
        bn_relu_kernel<<<512, 256, 0, stream>>>(Y, M, C, Cp, stats, g, beta);
    };

    // ---- Set-abstraction levels ----
    for (int l = 0; l < 4; ++l) {
        const int Nq = (l == 0) ? Nc : sa_np[l - 1];
        const float* xs = (l == 0) ? xyz : nxyz[l - 1];
        const float* fs = (l == 0) ? xyz : fslvl[l - 1];
        const int Cf = (l == 0) ? 3 : sa_mlp[l - 1][2];
        const int np = sa_np[l];
        const float r2 = sa_r[l] * sa_r[l];

        fps_kernel<<<Bc, 256, 0, stream>>>(xs, Nq, np, fps_idx, fps_dist);
        gather_xyz_kernel<<<ceil_div(Bc * np, 256), 256, 0, stream>>>(xs, fps_idx, Nq, np, Bc, nxyz[l]);
        ball_query_kernel<<<Bc * np, 128, 0, stream>>>(xs, nxyz[l], Nq, np, NS, r2, ball);

        const int M  = Bc * np * NS;
        const int K0 = 3 + Cf;
        const int Kp = pad64(K0);
        group_kernel<<<ceil_div(M, 256), 256, 0, stream>>>(xs, fs, nxyz[l], ball,
                                                           Nq, np, NS, Cf, Kp, M, gbuf);
        const float* cur = gbuf;
        int Ka = Kp, Kreal = K0;
        int CpLast = 0;
        for (int j = 0; j < 3; ++j) {
            const int Cout = sa_mlp[l][j];
            const int Cp   = pad64(Cout);
            float* out = (j == 1) ? actB : actA;
            run_gemm(cur, Ka, saw[l][j], Cout, Kreal, sab[l][j], out, Cp, M);
            run_bn(out, (long long)M, Cout, Cp, sag[l][j], sabt[l][j]);
            cur = out; Ka = Cp; Kreal = Cout; CpLast = Cp;
        }
        maxpool_kernel<<<ceil_div(Bc * np * Kreal, 256), 256, 0, stream>>>(
            cur, Bc * np, NS, Kreal, CpLast, fslvl[l]);
    }

    // ---- Feature propagation into fused buffer (part of d_out) ----
    float* out_x = (float*)d_out;
    float* fused = out_x + (size_t)Bc * Nc * 3;
    const int Ctot = 960;                 // multiple of 64 -> directly GEMM-ready
    int coff = 0;
    for (int l = 0; l < 4; ++l) {
        const int np = sa_np[l], Cs = sa_mlp[l][2];
        dim3 grid(ceil_div(Nc, 256), Bc);
        upsample_kernel<<<grid, 256, (size_t)np * 3 * 4, stream>>>(
            xyz, nxyz[l], fslvl[l], Nc, np, Cs, fused, Ctot, coff);
        coff += Cs;
    }

    // ---- Head MLP: 960 -> 512 -> 256 -> 128 -> 3 ----
    const int Mh = Bc * Nc;
    run_gemm(fused, 960, hw[0], 512, 960, hb[0], actA, 512, Mh);
    run_bn(actA, (long long)Mh, 512, 512, hg[0], hbt[0]);
    run_gemm(actA, 512, hw[1], 256, 512, hb[1], actB, 256, Mh);
    run_bn(actB, (long long)Mh, 256, 256, hg[1], hbt[1]);
    run_gemm(actB, 256, hw[2], 128, 256, hb[2], actA, 128, Mh);
    run_bn(actA, (long long)Mh, 128, 128, hg[2], hbt[2]);
    run_gemm(actA, 128, hw[3], 3, 128, hb[3], out_x, 3, Mh);   // final: bias only
}
